// MatchingNetwork_19662360281417
// MI455X (gfx1250) — compile-verified, matrix operand order checked by reading, not run on hardware
//
#include <hip/hip_runtime.h>
#include <hip/hip_bf16.h>

// ---- problem constants ----
#define BB 32
#define NN 64
#define KS 8
#define TT 64
#define LL 32
#define VV 32000
#define EE 64
#define HH 32
#define SS (NN*KS)   // 512
#define NSTEPS 5
#define ROWS (BB*TT) // 2048

typedef __attribute__((ext_vector_type(16))) _Float16 v16h;
typedef __attribute__((ext_vector_type(8)))  float    v8f;
typedef __attribute__((ext_vector_type(4)))  float    f32x4;
typedef __attribute__((ext_vector_type(4)))  unsigned u32x4;
typedef __attribute__((ext_vector_type(4)))  int      i32x4;
typedef __attribute__((ext_vector_type(8)))  int      i32x8;

#if defined(__has_builtin)
#if __has_builtin(__builtin_amdgcn_tensor_load_to_lds)
#define HAVE_TDM 1
#else
#define HAVE_TDM 0
#endif
#else
#define HAVE_TDM 0
#endif

__device__ __forceinline__ float rcp_fast(float x) { return __builtin_amdgcn_rcpf(x); }
// v_exp_f32 + v_rcp_f32, no IEEE div expansion
__device__ __forceinline__ float sigf(float x) {
  return rcp_fast(1.0f + __expf(-x));
}
__device__ __forceinline__ float tanh_fast(float x) {
  float e = __expf(2.0f * x);
  return 1.0f - 2.0f * rcp_fast(e + 1.0f);
}

// ---------------- WMMA fragment helpers (16x16x32 f16 -> f32) ----------------
__device__ __forceinline__ v16h load_a_frag(const _Float16* A, int lda,
                                            int mbase, int kbase, int lane) {
  int m  = mbase + (lane & 15);
  int kh = ((lane >> 4) & 1) * 8;
  const _Float16* p = A + (size_t)m * lda + kbase;
  v16h a;
#pragma unroll
  for (int j = 0; j < 8; ++j) { a[j] = p[kh + j]; a[8 + j] = p[16 + kh + j]; }
  return a;
}

__device__ __forceinline__ v16h load_b_frag(const _Float16* Bt, int ldb,
                                            int kbase, int nbase, int lane) {
  int n  = nbase + (lane & 15);
  int kh = ((lane >> 4) & 1) * 8;
  v16h b;
#pragma unroll
  for (int j = 0; j < 8; ++j) {
    b[j]     = Bt[(size_t)(kbase + kh + j)      * ldb + n];
    b[8 + j] = Bt[(size_t)(kbase + 16 + kh + j) * ldb + n];
  }
  return b;
}

__device__ __forceinline__ v16h load_bT_frag(const _Float16* Bn, int ldb,
                                             int kbase, int nbase, int lane) {
  int n  = nbase + (lane & 15);
  int kh = ((lane >> 4) & 1) * 8;
  const _Float16* p = Bn + (size_t)n * ldb + kbase;
  v16h b;
#pragma unroll
  for (int j = 0; j < 8; ++j) { b[j] = p[kh + j]; b[8 + j] = p[16 + kh + j]; }
  return b;
}

__device__ __forceinline__ void store_d_frag(float* D, int ldd, int mbase,
                                             int nbase, int lane, v8f d) {
  int n  = nbase + (lane & 15);
  int mo = (lane >> 4) & 1 ? 8 : 0;
#pragma unroll
  for (int r = 0; r < 8; ++r) D[(size_t)(mbase + r + mo) * ldd + n] = d[r];
}

__device__ __forceinline__ v8f wmma16(v16h a, v16h b, v8f c) {
  return __builtin_amdgcn_wmma_f32_16x16x32_f16(false, a, false, b, (short)0, c,
                                                false, false);
}

// ---------------- TDM helpers ----------------
#if HAVE_TDM
typedef __attribute__((address_space(3))) void lds_void;
__device__ __forceinline__ unsigned lds_addr_of(void* p) {
  return (unsigned)(unsigned long long)(lds_void*)p;
}
// 1-D contiguous f16 tile (nelem elements) Global -> LDS via Tensor Data Mover
__device__ __forceinline__ void tdm_load_1d_f16(unsigned lds_off,
                                                const _Float16* gsrc,
                                                unsigned nelem) {
  unsigned long long ga = (unsigned long long)gsrc;
  u32x4 g0;
  g0[0] = 1u;                                       // count=1, user mode
  g0[1] = lds_off;                                  // lds_addr
  g0[2] = (unsigned)ga;                             // global_addr[31:0]
  g0[3] = (unsigned)((ga >> 32) & 0x01ffffffu)      // global_addr[56:32]
        | (2u << 30);                               // type = 2 (image)
  i32x8 g1;
  g1[0] = (int)(1u << 16);                          // data_size = 1 -> 2 bytes
  g1[1] = (int)((nelem & 0xffffu) << 16);           // tensor_dim0[15:0]
  g1[2] = (int)(((nelem >> 16) & 0xffffu))          // tensor_dim0[31:16]
        | (1 << 16);                                // tensor_dim1 = 1
  g1[3] = (int)((nelem & 0xffffu) << 16);           // tile_dim0
  g1[4] = 0;                                        // tile_dim1/2 unused
  g1[5] = (int)nelem;                               // tensor_dim0_stride lo
  g1[6] = 0;
  g1[7] = 0;
  i32x4 z4 = {0, 0, 0, 0};
#if __clang_major__ >= 23
  i32x8 z8 = {0, 0, 0, 0, 0, 0, 0, 0};
  __builtin_amdgcn_tensor_load_to_lds(g0, g1, z4, z4, z8, 0);
#else
  __builtin_amdgcn_tensor_load_to_lds(g0, g1, z4, z4, 0);
#endif
}
__device__ __forceinline__ void wait_tensor0() {
#if __has_builtin(__builtin_amdgcn_s_wait_tensorcnt)
  __builtin_amdgcn_s_wait_tensorcnt(0);
#else
  asm volatile("s_wait_tensorcnt 0x0" ::: "memory");
#endif
}
#endif

// ---------------- small block-of-64 reductions ----------------
__device__ __forceinline__ float red64_max(float v, float* red, int e) {
  red[e] = v; __syncthreads();
#pragma unroll
  for (int s = 32; s >= 1; s >>= 1) {
    if (e < s) red[e] = fmaxf(red[e], red[e + s]);
    __syncthreads();
  }
  float r = red[0]; __syncthreads(); return r;
}
__device__ __forceinline__ float red64_sum(float v, float* red, int e) {
  red[e] = v; __syncthreads();
#pragma unroll
  for (int s = 32; s >= 1; s >>= 1) {
    if (e < s) red[e] += red[e + s];
    __syncthreads();
  }
  float r = red[0]; __syncthreads(); return r;
}

// ---------------- kernels ----------------

// streaming zero-fill, non-temporal (262 MB output > 192 MB L2)
__global__ void k_fill0(f32x4* p, long n4) {
  long i = (long)blockIdx.x * blockDim.x + threadIdx.x;
  long stride = (long)gridDim.x * blockDim.x;
  f32x4 z = {0.f, 0.f, 0.f, 0.f};
  for (; i < n4; i += stride) __builtin_nontemporal_store(z, p + i);
}

__global__ __launch_bounds__(256) void k_prep_weights(
    const float* W_ih, const float* W_hh, const float* b_ih, const float* b_hh,
    const float* fW_ih, const float* fW_hh, const float* fb_ih, const float* fb_hh,
    const float* bW_ih, const float* bW_hh, const float* bb_ih, const float* bb_hh,
    _Float16* wih_t, _Float16* whh_t, float* bias,
    _Float16* fwih_t, _Float16* fwhh_t, float* fbias,
    _Float16* bwih_t, _Float16* bwhh_t, float* bbias) {
  int i = blockIdx.x * blockDim.x + threadIdx.x;
  if (i < 16384) {
    int g = i >> 6, e = i & 63;
    wih_t[e * 256 + g] = (_Float16)W_ih[i];
    whh_t[e * 256 + g] = (_Float16)W_hh[i];
  }
  if (i < 8192) {
    int g = i >> 6, e = i & 63;
    fwih_t[e * 128 + g] = (_Float16)fW_ih[i];
    bwih_t[e * 128 + g] = (_Float16)bW_ih[i];
  }
  if (i < 4096) {
    int g = i >> 5, h = i & 31;
    fwhh_t[h * 128 + g] = (_Float16)fW_hh[i];
    bwhh_t[h * 128 + g] = (_Float16)bW_hh[i];
  }
  if (i < 256) bias[i] = b_ih[i] + b_hh[i];
  if (i < 128) { fbias[i] = fb_ih[i] + fb_hh[i]; bbias[i] = bb_ih[i] + bb_hh[i]; }
}

__global__ __launch_bounds__(64) void k_pool_support(const int* __restrict__ sup,
                                                     const float* __restrict__ emb,
                                                     _Float16* __restrict__ xs_h) {
  int r = blockIdx.x;            // r = b*S + s
  int e = threadIdx.x;
  int b = r >> 9, s = r & (SS - 1);
  const int* tok = sup + (size_t)r * LL;
  float acc = 0.f;
#pragma unroll 4
  for (int l = 0; l < LL; ++l) {
    int idx = tok[l];
    if (l + 1 < LL && (e & 15) == 0)
      __builtin_prefetch(emb + (size_t)tok[l + 1] * EE, 0, 3);
    if (idx != 0) acc += emb[(size_t)idx * EE + e];
  }
  xs_h[((size_t)s * BB + b) * EE + e] = (_Float16)acc;
}

__global__ __launch_bounds__(64) void k_pool_target(const int* __restrict__ tgt,
                                                    const float* __restrict__ emb,
                                                    float* __restrict__ x,
                                                    _Float16* __restrict__ x_h) {
  int r = blockIdx.x;            // b*T + t
  int e = threadIdx.x;
  const int* tok = tgt + (size_t)r * LL;
  float acc = 0.f;
#pragma unroll 4
  for (int l = 0; l < LL; ++l) {
    int idx = tok[l];
    if (idx != 0) acc += emb[(size_t)idx * EE + e];
  }
  x[(size_t)r * EE + e]   = acc;
  x_h[(size_t)r * EE + e] = (_Float16)acc;
}

// one-workgroup-per-direction BiLSTM; h/c in LDS; TDM double-buffers x tiles
__global__ __launch_bounds__(256) void k_bilstm(const _Float16* __restrict__ xs_h,
                                                const _Float16* __restrict__ fwih_t,
                                                const _Float16* __restrict__ fwhh_t,
                                                const float* __restrict__ fbias,
                                                const _Float16* __restrict__ bwih_t,
                                                const _Float16* __restrict__ bwhh_t,
                                                const float* __restrict__ bbias,
                                                float* __restrict__ sup_emb) {
  __shared__ _Float16 h_s[BB * HH];
  __shared__ float    c_s[BB * HH];
  __shared__ float    g_s[BB * 4 * HH];
#if HAVE_TDM
  __shared__ _Float16 xbuf[2][BB * EE];   // 2 x 4 KB double buffer
#endif

  const int dir  = blockIdx.x;           // 0 fwd, 1 bwd
  const int tid  = threadIdx.x;
  const int lane = tid & 31;
  const int wave = tid >> 5;
  const _Float16* wih = dir ? bwih_t : fwih_t;  // [64][128]
  const _Float16* whh = dir ? bwhh_t : fwhh_t;  // [32][128]
  const float*    bia = dir ? bbias  : fbias;

  for (int i = tid; i < BB * HH; i += 256) { h_s[i] = (_Float16)0.f; c_s[i] = 0.f; }

#if HAVE_TDM
  if (wave == 0) {
    int s0 = dir ? (SS - 1) : 0;
    tdm_load_1d_f16(lds_addr_of(&xbuf[0][0]), xs_h + (size_t)s0 * BB * EE, BB * EE);
    wait_tensor0();
  }
#endif
  __syncthreads();

  for (int step = 0; step < SS; ++step) {
    const int s = dir ? (SS - 1 - step) : step;
#if HAVE_TDM
    if (wave == 0 && step + 1 < SS) {   // async prefetch next step's x tile
      int s2 = dir ? (SS - 2 - step) : (step + 1);
      tdm_load_1d_f16(lds_addr_of(&xbuf[(step + 1) & 1][0]),
                      xs_h + (size_t)s2 * BB * EE, BB * EE);
    }
    const _Float16* xa = xbuf[step & 1];
#else
    const _Float16* xa = xs_h + (size_t)s * BB * EE;
#endif

    // gates[32][128] = x @ Wih^T + h @ Whh^T  (16 tiles; 8 waves x 2, unrolled)
#pragma unroll
    for (int q = 0; q < 2; ++q) {
      int tt = wave + q * 8;
      int mi = tt >> 3, ni = tt & 7;
      v8f acc = {};
#pragma unroll
      for (int kb = 0; kb < EE; kb += 32) {
        v16h a = load_a_frag(xa, EE, mi * 16, kb, lane);
        v16h b = load_b_frag(wih, 128, kb, ni * 16, lane);
        acc = wmma16(a, b, acc);
      }
      v16h ah = load_a_frag(h_s, HH, mi * 16, 0, lane);
      v16h bh = load_b_frag(whh, 128, 0, ni * 16, lane);
      acc = wmma16(ah, bh, acc);
      store_d_frag(g_s, 128, mi * 16, ni * 16, lane, acc);
    }
    __syncthreads();

    // elementwise cell: 32 rows x 32 hidden
#pragma unroll
    for (int it = 0; it < 4; ++it) {
      int i = tid + it * 256;
      int b = i >> 5, j = i & 31;
      float gi = g_s[b * 128 + j]      + bia[j];
      float gf = g_s[b * 128 + 32 + j] + bia[32 + j];
      float gg = g_s[b * 128 + 64 + j] + bia[64 + j];
      float go = g_s[b * 128 + 96 + j] + bia[96 + j];
      float c2 = sigf(gf) * c_s[b * HH + j] + sigf(gi) * tanh_fast(gg);
      float hh = sigf(go) * tanh_fast(c2);
      c_s[b * HH + j] = c2;
      h_s[b * HH + j] = (_Float16)hh;
      sup_emb[((size_t)b * SS + s) * EE + dir * HH + j] = hh;  // [B][S][E]
    }
#if HAVE_TDM
    if (wave == 0) wait_tensor0();      // next x tile resident before reuse
#endif
    __syncthreads();
  }
}

__global__ void k_hplusr(float* h, const float* r, _Float16* h_h, int n) {
  int i = blockIdx.x * blockDim.x + threadIdx.x;
  if (i < n) { float v = h[i] + r[i]; h[i] = v; h_h[i] = (_Float16)v; }
}

// gates[2048][256] = x @ W_ih^T + h @ W_hh^T + bias  (WMMA)
__global__ __launch_bounds__(256) void k_flayer_gemm(const _Float16* __restrict__ x_h,
                                                     const _Float16* __restrict__ h_h,
                                                     const _Float16* __restrict__ wih_t,
                                                     const _Float16* __restrict__ whh_t,
                                                     const float* __restrict__ bias,
                                                     float* __restrict__ gates) {
  const int lane = threadIdx.x & 31;
  const int wave = threadIdx.x >> 5;
  const int tile = blockIdx.x * 8 + wave;   // 2048 tiles = 128(M) x 16(N)
  const int mi = tile >> 4, ni = tile & 15;
  v8f acc = {};
#pragma unroll
  for (int kb = 0; kb < EE; kb += 32) {
    v16h a = load_a_frag(x_h, EE, mi * 16, kb, lane);
    v16h b = load_b_frag(wih_t, 256, kb, ni * 16, lane);
    acc = wmma16(a, b, acc);
    v16h a2 = load_a_frag(h_h, EE, mi * 16, kb, lane);
    v16h b2 = load_b_frag(whh_t, 256, kb, ni * 16, lane);
    acc = wmma16(a2, b2, acc);
  }
  float bv = bias[ni * 16 + (lane & 15)];
#pragma unroll
  for (int q = 0; q < 8; ++q) acc[q] += bv;
  store_d_frag(gates, 256, mi * 16, ni * 16, lane, acc);
}

__global__ void k_cell(const float* __restrict__ gates, const float* __restrict__ c,
                       const float* __restrict__ x, float* __restrict__ hn,
                       float* __restrict__ cn, int n) {
  int i = blockIdx.x * blockDim.x + threadIdx.x;
  if (i >= n) return;
  int row = i >> 6, e = i & 63;
  const float* g = gates + (size_t)row * 256;
  float gi = g[e], gf = g[64 + e], gg = g[128 + e], go = g[192 + e];
  float c2 = sigf(gf) * c[i] + sigf(gi) * tanh_fast(gg);
  cn[i] = c2;
  hn[i] = sigf(go) * tanh_fast(c2) + x[i];
}

__global__ void k_hsum(const float* __restrict__ h, float* __restrict__ hsum) {
  int i = blockIdx.x * blockDim.x + threadIdx.x;
  if (i >= BB * EE) return;
  int b = i >> 6, e = i & 63;
  float acc = 0.f;
#pragma unroll 8
  for (int t = 0; t < TT; ++t) acc += h[((size_t)(b * TT + t)) * EE + e];
  hsum[i] = acc;
}

__global__ __launch_bounds__(64) void k_attention(const float* __restrict__ hsum,
                                                  const float* __restrict__ sup_emb,
                                                  float* __restrict__ r) {
  __shared__ float red[64];
  int b = blockIdx.x >> 6, n = blockIdx.x & 63, e = threadIdx.x;
  float hs = hsum[b * EE + e];
  float racc = 0.f;
  for (int k = 0; k < KS; ++k) {
    const float* sup = sup_emb + ((size_t)(b * NN + n) * KS + k) * EE;
    float sv = sup[e];
    float m  = hs * sv;
    float mx = red64_max(m, red, e);
    float ex = __expf(m - mx);
    float sm = red64_sum(ex, red, e);
    racc += ex * rcp_fast(sm) * sv;
  }
  r[((size_t)(b * NN + n)) * EE + e] = racc;
}

__global__ __launch_bounds__(64) void k_tnorm(const float* __restrict__ h,
                                              _Float16* __restrict__ tn_h) {
  __shared__ float red[64];
  int row = blockIdx.x, e = threadIdx.x;
  float v = h[(size_t)row * EE + e];
  float ss = red64_sum(v * v, red, e);
  float inv = rcp_fast(fmaxf(sqrtf(ss), 1e-8f));
  tn_h[(size_t)row * EE + e] = (_Float16)(v * inv);
}

__global__ __launch_bounds__(64) void k_snorm(const float* __restrict__ sup_emb,
                                              _Float16* __restrict__ snk_h) {
  __shared__ float red[64];
  int b = blockIdx.x >> 6, n = blockIdx.x & 63, e = threadIdx.x;
  float acc = 0.f;
  for (int k = 0; k < KS; ++k) {
    const float* sup = sup_emb + ((size_t)(b * NN + n) * KS + k) * EE;
    float v = sup[e];
    float ss = red64_sum(v * v, red, e);
    float inv = rcp_fast(fmaxf(sqrtf(ss), 1e-8f));
    acc += v * inv;
  }
  snk_h[((size_t)(b * NN + n)) * EE + e] = (_Float16)acc;
}

__global__ __launch_bounds__(256) void k_sims(const _Float16* __restrict__ tn_h,
                                              const _Float16* __restrict__ snk_h,
                                              float* __restrict__ sims) {
  const int b = blockIdx.x;
  const int lane = threadIdx.x & 31;
  const int wave = threadIdx.x >> 5;
  const _Float16* A  = tn_h  + (size_t)b * TT * EE;   // [64][64]
  const _Float16* Bn = snk_h + (size_t)b * NN * EE;   // (k,n) = Bn[n][k]
  float* D = sims + (size_t)b * TT * NN;
#pragma unroll
  for (int q = 0; q < 2; ++q) {
    int tt = wave + q * 8;
    int mi = tt >> 2, ni = tt & 3;
    v8f acc = {};
#pragma unroll
    for (int kb = 0; kb < EE; kb += 32) {
      v16h a = load_a_frag(A, EE, mi * 16, kb, lane);
      v16h bf = load_bT_frag(Bn, EE, kb, ni * 16, lane);
      acc = wmma16(a, bf, acc);
    }
    store_d_frag(D, NN, mi * 16, ni * 16, lane, acc);
  }
}

__global__ __launch_bounds__(64) void k_scatter(const float* __restrict__ sims,
                                                const int* __restrict__ labels,
                                                float* __restrict__ out) {
  __shared__ float red[64];
  int row = blockIdx.x;      // b*T + t
  int e = threadIdx.x;       // n
  float v = sims[(size_t)row * NN + e];
  float mx = red64_max(v, red, e);
  float ex = __expf(v - mx);
  float den = red64_sum(ex, red, e);
  float asum = red64_sum(ex * rcp_fast(den), red, e);
  if (e == 0) out[(size_t)row * VV + labels[row]] = asum;
}

// ---------------- host launcher ----------------
extern "C" void kernel_launch(void* const* d_in, const int* in_sizes, int n_in,
                              void* d_out, int out_size, void* d_ws, size_t ws_size,
                              hipStream_t stream) {
  (void)in_sizes; (void)n_in; (void)ws_size;
  const int*   support = (const int*)d_in[0];
  const int*   target  = (const int*)d_in[1];
  const int*   labels  = (const int*)d_in[2];
  const float* emb     = (const float*)d_in[3];
  const float* W_ih    = (const float*)d_in[4];
  const float* W_hh    = (const float*)d_in[5];
  const float* b_ih    = (const float*)d_in[6];
  const float* b_hh    = (const float*)d_in[7];
  const float* fW_ih   = (const float*)d_in[8];
  const float* fW_hh   = (const float*)d_in[9];
  const float* fb_ih   = (const float*)d_in[10];
  const float* fb_hh   = (const float*)d_in[11];
  const float* bW_ih   = (const float*)d_in[12];
  const float* bW_hh   = (const float*)d_in[13];
  const float* bb_ih   = (const float*)d_in[14];
  const float* bb_hh   = (const float*)d_in[15];

  char* ws = (char*)d_ws;
  size_t off = 0;
  auto alloc = [&](size_t bytes) -> char* {
    char* p = ws + off;
    off += (bytes + 255) & ~(size_t)255;
    return p;
  };

  _Float16* xs_h    = (_Float16*)alloc((size_t)SS * BB * EE * 2);
  float*    sup_emb = (float*)   alloc((size_t)BB * SS * EE * 4);
  float*    x       = (float*)   alloc((size_t)ROWS * EE * 4);
  _Float16* x_h     = (_Float16*)alloc((size_t)ROWS * EE * 2);
  float*    h       = (float*)   alloc((size_t)ROWS * EE * 4);   // h,c,r contiguous
  float*    c       = (float*)   alloc((size_t)ROWS * EE * 4);
  float*    r       = (float*)   alloc((size_t)ROWS * EE * 4);
  float*    hn      = (float*)   alloc((size_t)ROWS * EE * 4);
  float*    cn      = (float*)   alloc((size_t)ROWS * EE * 4);
  _Float16* h_h     = (_Float16*)alloc((size_t)ROWS * EE * 2);
  float*    gates   = (float*)   alloc((size_t)ROWS * 256 * 4);
  float*    hsum    = (float*)   alloc((size_t)BB * EE * 4);
  _Float16* wih_t   = (_Float16*)alloc(16384 * 2);
  _Float16* whh_t   = (_Float16*)alloc(16384 * 2);
  float*    bias    = (float*)   alloc(256 * 4);
  _Float16* fwih_t  = (_Float16*)alloc(8192 * 2);
  _Float16* fwhh_t  = (_Float16*)alloc(4096 * 2);
  float*    fbias   = (float*)   alloc(128 * 4);
  _Float16* bwih_t  = (_Float16*)alloc(8192 * 2);
  _Float16* bwhh_t  = (_Float16*)alloc(4096 * 2);
  float*    bbias   = (float*)   alloc(128 * 4);
  _Float16* tn_h    = (_Float16*)alloc((size_t)ROWS * EE * 2);
  _Float16* snk_h   = (_Float16*)alloc((size_t)BB * NN * EE * 2);
  float*    sims    = (float*)   alloc((size_t)BB * TT * NN * 4);

  k_prep_weights<<<64, 256, 0, stream>>>(W_ih, W_hh, b_ih, b_hh,
                                         fW_ih, fW_hh, fb_ih, fb_hh,
                                         bW_ih, bW_hh, bb_ih, bb_hh,
                                         wih_t, whh_t, bias,
                                         fwih_t, fwhh_t, fbias,
                                         bwih_t, bwhh_t, bbias);
  k_pool_support<<<BB * SS, 64, 0, stream>>>(support, emb, xs_h);
  k_pool_target<<<ROWS, 64, 0, stream>>>(target, emb, x, x_h);
  k_fill0<<<256, 256, 0, stream>>>((f32x4*)h, (long)(3 * ROWS * EE / 4));
  k_bilstm<<<2, 256, 0, stream>>>(xs_h, fwih_t, fwhh_t, fbias,
                                  bwih_t, bwhh_t, bbias, sup_emb);
  float *hcur = h, *ccur = c, *hnext = hn, *cnext = cn;
  for (int it = 0; it < NSTEPS; ++it) {
    k_hplusr<<<ROWS * EE / 256, 256, 0, stream>>>(hcur, r, h_h, ROWS * EE);
    k_flayer_gemm<<<256, 256, 0, stream>>>(x_h, h_h, wih_t, whh_t, bias, gates);
    k_cell<<<ROWS * EE / 256, 256, 0, stream>>>(gates, ccur, x, hnext, cnext,
                                                ROWS * EE);
    k_hsum<<<8, 256, 0, stream>>>(hcur, hsum);
    k_attention<<<BB * NN, 64, 0, stream>>>(hsum, sup_emb, r);
    float* tp = hcur; hcur = hnext; hnext = tp;
    tp = ccur; ccur = cnext; cnext = tp;
  }
  k_tnorm<<<ROWS, 64, 0, stream>>>(hcur, tn_h);
  k_snorm<<<BB * NN, 64, 0, stream>>>(sup_emb, snk_h);
  k_sims<<<BB, 256, 0, stream>>>(tn_h, snk_h, sims);
  long n4 = (long)out_size / 4;
  k_fill0<<<8192, 256, 0, stream>>>((f32x4*)d_out, n4);
  k_scatter<<<ROWS, 64, 0, stream>>>(sims, labels, (float*)d_out);
}
